// GC_VAE_35227321761815
// MI455X (gfx1250) — matrix operand order repaired and checked by reading, not compiled, not run on hardware
//
#include <hip/hip_runtime.h>
#include <hip/hip_bf16.h>
#include <math.h>

// ---------------------------------------------------------------------------
// Problem constants (from reference)
// ---------------------------------------------------------------------------
#define NNODES 8192
#define NEDGES 262144
#define NFEAT  1024
#define NHID   512
#define ZDIM   32

typedef __bf16 bf16_t;
typedef __attribute__((ext_vector_type(16))) __bf16 v16bf;
typedef __attribute__((ext_vector_type(8)))  __bf16 v8bf;
typedef __attribute__((ext_vector_type(8)))  float  v8f;

// ---------------------------------------------------------------------------
// fp32 -> bf16 convert (grid-stride)
// ---------------------------------------------------------------------------
__global__ void k_cvt_bf16(const float* __restrict__ in, bf16_t* __restrict__ out, int n) {
    for (int i = blockIdx.x * blockDim.x + threadIdx.x; i < n; i += gridDim.x * blockDim.x)
        out[i] = (bf16_t)in[i];
}

// fp32 [K,N] row-major -> bf16 [N,K] (column-major weight, contiguous-in-K)
__global__ void k_cvt_transpose_bf16(const float* __restrict__ W, bf16_t* __restrict__ Wt,
                                     int K, int N) {
    int idx = blockIdx.x * blockDim.x + threadIdx.x;   // idx over K*N, input-coalesced
    if (idx >= K * N) return;
    int k = idx / N, n = idx % N;
    Wt[(size_t)n * K + k] = (bf16_t)W[idx];
}

// ---------------------------------------------------------------------------
// WMMA bf16 GEMM:  C[M,N] (f32, row-major) = A[M,K] (bf16, row-major)
//                                           @ Bt[N,K] (bf16, K-contiguous)
// One wave -> one 16x16 tile. 8 waves / block. K multiple of 32.
// Fragment layouts per CDNA5 ISA 7.12.2.
// ---------------------------------------------------------------------------
__global__ void k_gemm_bf16_wmma(const bf16_t* __restrict__ A,
                                 const bf16_t* __restrict__ Bt,
                                 float* __restrict__ C,
                                 int M, int N, int K) {
    const int wave   = threadIdx.x >> 5;
    const int lane   = threadIdx.x & 31;
    const int half   = lane >> 4;      // 0: lanes 0-15, 1: lanes 16-31
    const int lr     = lane & 15;
    const int tilesN = N >> 4;
    const int tileId = blockIdx.x * 8 + wave;
    if (tileId >= (M >> 4) * tilesN) return;           // wave-uniform guard
    const int mBase  = (tileId / tilesN) << 4;
    const int nBase  = (tileId % tilesN) << 4;

    const bf16_t* Arow = A  + (size_t)(mBase + lr) * K;        // A: row = mBase+lr
    const bf16_t* Bcol = Bt + (size_t)(nBase + lr) * K;        // B: col = nBase+lr

    v8f acc = {};
    for (int kb = 0; kb < K; kb += 32) {
        // A 16x32 bf16: e0..7 -> K = kb + half*8 + e ; e8..15 -> K = kb+16 + half*8 + (e-8)
        v8bf alo = *(const v8bf*)(Arow + kb + half * 8);
        v8bf ahi = *(const v8bf*)(Arow + kb + 16 + half * 8);
        v16bf a  = __builtin_shufflevector(alo, ahi,
                      0, 1, 2, 3, 4, 5, 6, 7, 8, 9, 10, 11, 12, 13, 14, 15);
        // B 32x16 bf16: e0..7 -> K = kb + half*8 + e ; e8..15 -> K = kb+16 + half*8 + (e-8)
        v8bf blo = *(const v8bf*)(Bcol + kb + half * 8);
        v8bf bhi = *(const v8bf*)(Bcol + kb + 16 + half * 8);
        v16bf b  = __builtin_shufflevector(blo, bhi,
                      0, 1, 2, 3, 4, 5, 6, 7, 8, 9, 10, 11, 12, 13, 14, 15);

        acc = __builtin_amdgcn_wmma_f32_16x16x32_bf16(
                  false, a, false, b, (short)0, acc, false, false);

        if (kb + 64 <= K) {                  // prefetch next-next K slab of A
            __builtin_prefetch(Arow + kb + 64, 0, 1);
        }
    }

    // C/D 16x16 f32: vgpr r -> row mBase + half*8 + r, lane lr -> col nBase+lr
    float* Crow = C + (size_t)(mBase + half * 8) * N + nBase + lr;
#pragma unroll
    for (int r = 0; r < 8; ++r)
        Crow[(size_t)r * N] = acc[r];
}

// ---------------------------------------------------------------------------
// Decoder: adj_recon = sigmoid(z @ z^T), z bf16 [NNODES, 32]. K=32 -> 1 WMMA/tile.
// ---------------------------------------------------------------------------
__global__ void k_decoder_wmma(const bf16_t* __restrict__ Z,
                               float* __restrict__ out) {
    const int wave   = threadIdx.x >> 5;
    const int lane   = threadIdx.x & 31;
    const int half   = lane >> 4;
    const int lr     = lane & 15;
    const int tilesN = NNODES >> 4;                     // 512
    const int tileId = blockIdx.x * 8 + wave;
    const int mBase  = (tileId / tilesN) << 4;
    const int nBase  = (tileId % tilesN) << 4;

    const bf16_t* Arow = Z + (size_t)(mBase + lr) * ZDIM;
    const bf16_t* Bcol = Z + (size_t)(nBase + lr) * ZDIM; // z^T: B[k][n] = z[n][k]

    v8bf alo = *(const v8bf*)(Arow + half * 8);
    v8bf ahi = *(const v8bf*)(Arow + 16 + half * 8);
    v16bf a  = __builtin_shufflevector(alo, ahi,
                  0, 1, 2, 3, 4, 5, 6, 7, 8, 9, 10, 11, 12, 13, 14, 15);
    v8bf blo = *(const v8bf*)(Bcol + half * 8);
    v8bf bhi = *(const v8bf*)(Bcol + 16 + half * 8);
    v16bf b  = __builtin_shufflevector(blo, bhi,
                  0, 1, 2, 3, 4, 5, 6, 7, 8, 9, 10, 11, 12, 13, 14, 15);

    v8f acc = {};
    acc = __builtin_amdgcn_wmma_f32_16x16x32_bf16(
              false, a, false, b, (short)0, acc, false, false);

    float* Crow = out + (size_t)(mBase + half * 8) * NNODES + nBase + lr;
#pragma unroll
    for (int r = 0; r < 8; ++r) {
        float s = 1.0f / (1.0f + __expf(-acc[r]));
        Crow[(size_t)r * NNODES] = s;
    }
}

// ---------------------------------------------------------------------------
// Row-range binary search over sorted adj_rows
// ---------------------------------------------------------------------------
__device__ __forceinline__ int lower_bound_i32(const int* __restrict__ arr, int n, int key) {
    int lo = 0, hi = n;
    while (lo < hi) {
        int mid = (lo + hi) >> 1;
        if (arr[mid] < key) lo = mid + 1; else hi = mid;
    }
    return lo;
}

// ---------------------------------------------------------------------------
// SpMM1: h = relu(spmm(adj, XW) + b0), output bf16 for next GEMM.
// One block (256 thr) per row; 2 feature columns per thread (D = 512).
// ---------------------------------------------------------------------------
__global__ void k_spmm_bias_relu(const int* __restrict__ rows,
                                 const int* __restrict__ cols,
                                 const float* __restrict__ vals,
                                 const float* __restrict__ XW,   // [NNODES, NHID] f32
                                 const float* __restrict__ bias, // [NHID]
                                 bf16_t* __restrict__ Hout) {    // [NNODES, NHID] bf16
    const int row = blockIdx.x;
    const int lo  = lower_bound_i32(rows, NEDGES, row);
    const int hi  = lower_bound_i32(rows, NEDGES, row + 1);

    const int c0 = threadIdx.x;
    const int c1 = threadIdx.x + 256;
    float a0 = 0.0f, a1 = 0.0f;
    for (int e = lo; e < hi; ++e) {
        const float w = vals[e];
        const float* src = XW + (size_t)cols[e] * NHID;
        a0 += w * src[c0];
        a1 += w * src[c1];
    }
    a0 = fmaxf(a0 + bias[c0], 0.0f);
    a1 = fmaxf(a1 + bias[c1], 0.0f);
    bf16_t* dst = Hout + (size_t)row * NHID;
    dst[c0] = (bf16_t)a0;
    dst[c1] = (bf16_t)a1;
}

// ---------------------------------------------------------------------------
// SpMM2 (fused heads): mu = relu(spmm(adj, HW1)+b1), logvar = relu(spmm(adj,HW2)+b2)
// z = mu. 64 threads/row: lanes 0-31 -> mu head, 32-63 -> logvar head.
// ---------------------------------------------------------------------------
__global__ void k_spmm2_heads(const int* __restrict__ rows,
                              const int* __restrict__ cols,
                              const float* __restrict__ vals,
                              const float* __restrict__ HW1,   // [NNODES, 32]
                              const float* __restrict__ HW2,   // [NNODES, 32]
                              const float* __restrict__ b1,
                              const float* __restrict__ b2,
                              float* __restrict__ z_out,
                              float* __restrict__ mu_out,
                              float* __restrict__ logvar_out,
                              bf16_t* __restrict__ z_bf16) {
    const int row  = blockIdx.x;
    const int lo   = lower_bound_i32(rows, NEDGES, row);
    const int hi   = lower_bound_i32(rows, NEDGES, row + 1);
    const int head = threadIdx.x >> 5;        // 0 = mu, 1 = logvar
    const int d    = threadIdx.x & 31;
    const float* HW = head ? HW2 : HW1;

    float acc = 0.0f;
    for (int e = lo; e < hi; ++e)
        acc += vals[e] * HW[(size_t)cols[e] * ZDIM + d];

    const size_t o = (size_t)row * ZDIM + d;
    if (head == 0) {
        float v = fmaxf(acc + b1[d], 0.0f);
        z_out[o]  = v;                 // z = mu (eval-mode reparameterize)
        mu_out[o] = v;
        z_bf16[o] = (bf16_t)v;
    } else {
        float v = fmaxf(acc + b2[d], 0.0f);
        logvar_out[o] = v;
    }
}

// ---------------------------------------------------------------------------
// Launch
// ---------------------------------------------------------------------------
extern "C" void kernel_launch(void* const* d_in, const int* in_sizes, int n_in,
                              void* d_out, int out_size, void* d_ws, size_t ws_size,
                              hipStream_t stream) {
    const float* x        = (const float*)d_in[0];
    const float* adj_vals = (const float*)d_in[1];
    const float* W0       = (const float*)d_in[2];
    const float* b0       = (const float*)d_in[3];
    const float* W1       = (const float*)d_in[4];
    const float* b1       = (const float*)d_in[5];
    const float* W2       = (const float*)d_in[6];
    const float* b2       = (const float*)d_in[7];
    const int*   adj_rows = (const int*)d_in[8];
    const int*   adj_cols = (const int*)d_in[9];

    float* out        = (float*)d_out;
    float* adj_recon  = out;                                   // [N, N]
    float* z_out      = adj_recon + (size_t)NNODES * NNODES;   // [N, 32]
    float* mu_out     = z_out + (size_t)NNODES * ZDIM;
    float* logvar_out = mu_out + (size_t)NNODES * ZDIM;

    // --- workspace carve-up (bf16 staging + f32 intermediates), ~44 MB ---
    char* ws = (char*)d_ws;
    bf16_t* x_bf16  = (bf16_t*)ws;  ws += (size_t)NNODES * NFEAT * sizeof(bf16_t);
    bf16_t* W0t     = (bf16_t*)ws;  ws += (size_t)NFEAT * NHID * sizeof(bf16_t);
    bf16_t* W1t     = (bf16_t*)ws;  ws += (size_t)NHID * ZDIM * sizeof(bf16_t);
    bf16_t* W2t     = (bf16_t*)ws;  ws += (size_t)NHID * ZDIM * sizeof(bf16_t);
    float*  xW0     = (float*)ws;   ws += (size_t)NNODES * NHID * sizeof(float);
    bf16_t* h_bf16  = (bf16_t*)ws;  ws += (size_t)NNODES * NHID * sizeof(bf16_t);
    float*  hW1     = (float*)ws;   ws += (size_t)NNODES * ZDIM * sizeof(float);
    float*  hW2     = (float*)ws;   ws += (size_t)NNODES * ZDIM * sizeof(float);
    bf16_t* z_bf16  = (bf16_t*)ws;  ws += (size_t)NNODES * ZDIM * sizeof(bf16_t);

    // 1) precision staging
    k_cvt_bf16<<<4096, 256, 0, stream>>>(x, x_bf16, NNODES * NFEAT);
    k_cvt_transpose_bf16<<<(NFEAT * NHID + 255) / 256, 256, 0, stream>>>(W0, W0t, NFEAT, NHID);
    k_cvt_transpose_bf16<<<(NHID * ZDIM + 255) / 256, 256, 0, stream>>>(W1, W1t, NHID, ZDIM);
    k_cvt_transpose_bf16<<<(NHID * ZDIM + 255) / 256, 256, 0, stream>>>(W2, W2t, NHID, ZDIM);

    // 2) xW0 = x @ W0   (8192x1024x512) -> 16384 tiles / 8 waves
    k_gemm_bf16_wmma<<<2048, 256, 0, stream>>>(x_bf16, W0t, xW0, NNODES, NHID, NFEAT);

    // 3) h = relu(spmm(adj, xW0) + b0)  (emit bf16)
    k_spmm_bias_relu<<<NNODES, 256, 0, stream>>>(adj_rows, adj_cols, adj_vals,
                                                 xW0, b0, h_bf16);

    // 4) hW1 / hW2  (8192x512x32) -> 1024 tiles / 8 waves each
    k_gemm_bf16_wmma<<<128, 256, 0, stream>>>(h_bf16, W1t, hW1, NNODES, ZDIM, NHID);
    k_gemm_bf16_wmma<<<128, 256, 0, stream>>>(h_bf16, W2t, hW2, NNODES, ZDIM, NHID);

    // 5) mu / logvar heads (fused), z = mu
    k_spmm2_heads<<<NNODES, 64, 0, stream>>>(adj_rows, adj_cols, adj_vals,
                                             hW1, hW2, b1, b2,
                                             z_out, mu_out, logvar_out, z_bf16);

    // 6) adj_recon = sigmoid(z @ z^T)  (512x512 tiles = 262144 / 8 waves)
    k_decoder_wmma<<<32768, 256, 0, stream>>>(z_bf16, adj_recon);
}
